// ImprovedGNNModel_86638080295546
// MI455X (gfx1250) — matrix operand order repaired and checked by reading, not compiled, not run on hardware
//
#include <hip/hip_runtime.h>

#define N_NODES 8192
#define N_EDGES 40960
#define FIN     32
#define EDIM    8
#define HDIM    64
#define GNUM    64
#define BN_EPS  1e-5f

typedef __attribute__((ext_vector_type(16))) __bf16 v16bf;
typedef __attribute__((ext_vector_type(8)))  float  v8f;

union ABu { uint4 q[2]; v16bf v; };

__device__ __forceinline__ unsigned short f32_to_bf16(float f) {
  unsigned u = __float_as_uint(f);
  u += 0x7FFFu + ((u >> 16) & 1u);           // round-to-nearest-even
  return (unsigned short)(u >> 16);
}

// ---------------- edge MLP: h = relu(edge_attr @ w1 + b1), store bf16 (E x 64)
__global__ void edge_mlp_kernel(const float* __restrict__ ea,
                                const float* __restrict__ w1,
                                const float* __restrict__ b1,
                                unsigned short* __restrict__ h_bf) {
  int t = blockIdx.x * blockDim.x + threadIdx.x;
  if (t >= N_EDGES * 64) return;
  int e = t >> 6, j = t & 63;
  const float* er = ea + (size_t)e * EDIM;
  float acc = b1[j];
#pragma unroll
  for (int d = 0; d < EDIM; ++d) acc += er[d] * w1[d * 64 + j];
  h_bf[t] = f32_to_bf16(fmaxf(acc, 0.f));
}

// ---------------- pack w2 (64 x NT*16, f32) into WMMA-B tile layout, bf16
// Wp[((T*2+ks)*32 + lane)*16 + e] = w2[ks*32 + (lane/16)*16 + e][T*16 + lane%16]
__global__ void pack_w2_kernel(const float* __restrict__ w2,
                               unsigned short* __restrict__ Wp, int NT) {
  int t = blockIdx.x * blockDim.x + threadIdx.x;
  if (t >= NT * 1024) return;
  int e    = t & 15;
  int lane = (t >> 4) & 31;
  int ks   = (t >> 9) & 1;
  int T    = t >> 10;
  int K = ks * 32 + ((lane >> 4) << 4) + e;
  int n = T * 16 + (lane & 15);
  Wp[t] = f32_to_bf16(w2[(size_t)K * (NT * 16) + n]);
}

// ---------------- in-degree counts
__global__ void count_kernel(const int* __restrict__ dst, float* __restrict__ cnt) {
  int e = blockIdx.x * blockDim.x + threadIdx.x;
  if (e < N_EDGES) atomicAdd(&cnt[dst[e]], 1.0f);
}

// ---------------- fused NNConv edge kernel (the hot path, WMMA bf16)
// one wave handles 16 edges: D = h(16x64) x w2-tile(64x16) via 2x wmma(K=32),
// then contracts with x[src] into a REGISTER accumulator:
// lane (g,l16) owns msg[r+8g][ot*16+l16] for r<8, ot<4 -> float acc[8][4].
#define EW 8   // waves per block
__global__ __launch_bounds__(32 * EW)
void nnconv_edge_kernel(const unsigned short* __restrict__ h_bf,
                        const unsigned short* __restrict__ Wp,
                        const float* __restrict__ b2,
                        const float* __restrict__ x,
                        const int* __restrict__ src,
                        const int* __restrict__ dst,
                        float* __restrict__ s,
                        int in_ch, int NT) {
  __shared__ float xs[EW][16 * 64];
  const int wave = threadIdx.x >> 5;
  const int lane = threadIdx.x & 31;
  const int g = lane >> 4, l16 = lane & 15;
  const int tile = blockIdx.x * EW + wave;
  const int eBase = tile * 16;

  for (int i = lane; i < 16 * in_ch; i += 32) {
    int r = i / in_ch, c = i - r * in_ch;
    xs[wave][r * in_ch + c] = x[(size_t)src[eBase + r] * in_ch + c];
  }

  // A operand: row = lane%16 (edge), K = ks*32 + 16*g + e
  ABu a0, a1;
  const uint4* hp = (const uint4*)(h_bf + (size_t)(eBase + l16) * 64);
  a0.q[0] = hp[2 * g];     a0.q[1] = hp[2 * g + 1];
  a1.q[0] = hp[4 + 2 * g]; a1.q[1] = hp[5 + 2 * g];

  float acc[8][4];
#pragma unroll
  for (int r = 0; r < 8; ++r)
#pragma unroll
    for (int ot = 0; ot < 4; ++ot) acc[r][ot] = 0.f;

  const int NT4 = NT >> 2;
  for (int T4 = 0; T4 < NT4; ++T4) {
    // i = T/4 is constant across the 4 inner tiles: hoist the x gathers
    float xv[8];
#pragma unroll
    for (int r = 0; r < 8; ++r) xv[r] = xs[wave][(r + 8 * g) * in_ch + T4];

#pragma unroll
    for (int ot = 0; ot < 4; ++ot) {
      const int T = T4 * 4 + ot;
      ABu b0, b1;
      const uint4* bp = (const uint4*)Wp + ((size_t)(T * 2) * 32 + lane) * 2;
      b0.q[0] = bp[0];  b0.q[1] = bp[1];
      b1.q[0] = bp[64]; b1.q[1] = bp[65];   // +32 lanes * 2 uint4 = next k-step
      v8f c = {};
      c = __builtin_amdgcn_wmma_f32_16x16x32_bf16(false, a0.v, false, b0.v,
                                                  (short)0, c, false, false);
      c = __builtin_amdgcn_wmma_f32_16x16x32_bf16(false, a1.v, false, b1.v,
                                                  (short)0, c, false, false);
      // D layout: elem r -> row r + 8*g, col = lane%16
#pragma unroll
      for (int r = 0; r < 8; ++r) acc[r][ot] += xv[r] * c[r];
    }
  }

  // bias part of We: msg[e][o] += sum_i x[src][i] * b2[i*64+o]
#pragma unroll
  for (int r = 0; r < 8; ++r) {
    const int row = r + 8 * g;
#pragma unroll
    for (int ot = 0; ot < 4; ++ot) {
      const int o = ot * 16 + l16;
      float a = 0.f;
      for (int i = 0; i < in_ch; ++i) a += xs[wave][row * in_ch + i] * b2[i * 64 + o];
      acc[r][ot] += a;
    }
  }

  // scatter-add each lane's 32 owned message values into the segment sum
#pragma unroll
  for (int r = 0; r < 8; ++r) {
    const int row = r + 8 * g;
    const size_t base = (size_t)dst[eBase + row] * 64 + l16;
#pragma unroll
    for (int ot = 0; ot < 4; ++ot)
      atomicAdd(&s[base + ot * 16], acc[r][ot]);
  }
}

// ---------------- mean aggregate + root transform
__global__ void finalize_kernel(const float* __restrict__ s,
                                const float* __restrict__ cnt,
                                const float* __restrict__ x,
                                const float* __restrict__ rw,
                                const float* __restrict__ rb,
                                float* __restrict__ pre, int in_ch) {
  int t = blockIdx.x * blockDim.x + threadIdx.x;
  if (t >= N_NODES * 64) return;
  int n = t >> 6, o = t & 63;
  float acc = s[t] / fmaxf(cnt[n], 1.0f) + rb[o];
  const float* xr = x + (size_t)n * in_ch;
  for (int i = 0; i < in_ch; ++i) acc += xr[i] * rw[i * 64 + o];
  pre[t] = acc;
}

// ---------------- batchnorm stats (column sums / sumsq over rows)
__global__ void bn_stats_kernel(const float* __restrict__ pre,
                                float* __restrict__ stats, int cols, int rows) {
  int col = threadIdx.x % cols;
  int rpb = blockDim.x / cols;
  int r0 = blockIdx.x * rpb + threadIdx.x / cols;
  int stride = gridDim.x * rpb;
  float sum = 0.f, sq = 0.f;
  for (int r = r0; r < rows; r += stride) {
    float v = pre[(size_t)r * cols + col];
    sum += v; sq += v * v;
  }
  atomicAdd(&stats[col], sum);
  atomicAdd(&stats[cols + col], sq);
}

__global__ void bn_apply_kernel(const float* __restrict__ pre,
                                const float* __restrict__ stats,
                                const float* __restrict__ gg,
                                const float* __restrict__ bb,
                                const float* __restrict__ residual,
                                float* __restrict__ out, int cols, int rows) {
  int t = blockIdx.x * blockDim.x + threadIdx.x;
  if (t >= rows * cols) return;
  int o = t % cols;
  float mean = stats[o] / rows;
  float var = stats[cols + o] / rows - mean * mean;
  float y = (pre[t] - mean) * rsqrtf(var + BN_EPS) * gg[o] + bb[o];
  y = fmaxf(y, 0.f);
  if (residual) y += residual[t];
  out[t] = y;
}

// ---------------- attention gate + segment max (monotonic-uint atomicMax)
__global__ __launch_bounds__(256)
void gate_kernel(const float* __restrict__ x3, const float* __restrict__ gw1,
                 const float* __restrict__ gb1, const float* __restrict__ gw2,
                 const float* __restrict__ gb2, float* __restrict__ gate,
                 unsigned* __restrict__ mu, const int* __restrict__ batch) {
  int node = blockIdx.x * 8 + (threadIdx.x >> 5);
  int j = threadIdx.x & 31;
  const float* xr = x3 + (size_t)node * 64;
  float acc = gb1[j];
  for (int i = 0; i < 64; ++i) acc += xr[i] * gw1[i * 32 + j];
  acc = fmaxf(acc, 0.f) * gw2[j];
  for (int off = 16; off; off >>= 1) acc += __shfl_down(acc, off, 32);
  if (j == 0) {
    float gt = acc + gb2[0];
    gate[node] = gt;
    unsigned u = __float_as_uint(gt);
    u = (u & 0x80000000u) ? ~u : (u | 0x80000000u);
    atomicMax(&mu[batch[node]], u);
  }
}

__global__ void attn_exp_kernel(const float* __restrict__ gate,
                                const unsigned* __restrict__ mu,
                                const int* __restrict__ batch,
                                float* __restrict__ eexp, float* __restrict__ z) {
  int n = blockIdx.x * blockDim.x + threadIdx.x;
  if (n >= N_NODES) return;
  unsigned u = mu[batch[n]];
  float m = __uint_as_float((u & 0x80000000u) ? (u & 0x7FFFFFFFu) : ~u);
  float e = expf(gate[n] - m);
  eexp[n] = e;
  atomicAdd(&z[batch[n]], e);
}

__global__ void attn_pool_kernel(const float* __restrict__ x3,
                                 const float* __restrict__ eexp,
                                 const float* __restrict__ z,
                                 const int* __restrict__ batch,
                                 float* __restrict__ p) {
  int t = blockIdx.x * blockDim.x + threadIdx.x;
  if (t >= N_NODES * 64) return;
  int n = t >> 6, o = t & 63;
  float alpha = eexp[n] / z[batch[n]];
  atomicAdd(&p[(size_t)batch[n] * 64 + o], alpha * x3[t]);
}

// ---------------- head: single block, each thread owns a full BN column
__global__ void head_fc_bn_relu(const float* __restrict__ in,
                                const float* __restrict__ W,
                                const float* __restrict__ b,
                                const float* __restrict__ gbn,
                                const float* __restrict__ bbn,
                                float* __restrict__ out, int I, int O) {
  int o = threadIdx.x;
  if (o >= O) return;
  float col[GNUM];
  float sum = 0.f, sq = 0.f;
  for (int gi = 0; gi < GNUM; ++gi) {
    const float* r = in + (size_t)gi * I;
    float acc = b[o];
    for (int i = 0; i < I; ++i) acc += r[i] * W[i * O + o];
    col[gi] = acc; sum += acc; sq += acc * acc;
  }
  float mean = sum / GNUM, var = sq / GNUM - mean * mean;
  float inv = rsqrtf(var + BN_EPS);
  for (int gi = 0; gi < GNUM; ++gi)
    out[(size_t)gi * O + o] = fmaxf((col[gi] - mean) * inv * gbn[o] + bbn[o], 0.f);
}

__global__ void head_fc3(const float* __restrict__ in, const float* __restrict__ w,
                         const float* __restrict__ b, float* __restrict__ out) {
  int gi = threadIdx.x;
  if (gi >= GNUM) return;
  float acc = b[0];
  const float* r = in + (size_t)gi * 64;
  for (int i = 0; i < 64; ++i) acc += r[i] * w[i];
  out[gi] = acc;
}

extern "C" void kernel_launch(void* const* d_in, const int* in_sizes, int n_in,
                              void* d_out, int out_size, void* d_ws, size_t ws_size,
                              hipStream_t stream) {
  const float* x   = (const float*)d_in[0];
  const int*   ei  = (const int*)d_in[1];
  const float* ea  = (const float*)d_in[2];
  const int*   bat = (const int*)d_in[3];
  const float* ew1[3] = {(const float*)d_in[4],  (const float*)d_in[12], (const float*)d_in[20]};
  const float* eb1[3] = {(const float*)d_in[5],  (const float*)d_in[13], (const float*)d_in[21]};
  const float* ew2[3] = {(const float*)d_in[6],  (const float*)d_in[14], (const float*)d_in[22]};
  const float* eb2[3] = {(const float*)d_in[7],  (const float*)d_in[15], (const float*)d_in[23]};
  const float* rw [3] = {(const float*)d_in[8],  (const float*)d_in[16], (const float*)d_in[24]};
  const float* rb [3] = {(const float*)d_in[9],  (const float*)d_in[17], (const float*)d_in[25]};
  const float* bg [3] = {(const float*)d_in[10], (const float*)d_in[18], (const float*)d_in[26]};
  const float* bb [3] = {(const float*)d_in[11], (const float*)d_in[19], (const float*)d_in[27]};
  const float* gw1 = (const float*)d_in[28]; const float* gb1 = (const float*)d_in[29];
  const float* gw2 = (const float*)d_in[30]; const float* gb2 = (const float*)d_in[31];
  const float* f1w = (const float*)d_in[32]; const float* f1b = (const float*)d_in[33];
  const float* f1g = (const float*)d_in[34]; const float* f1bb= (const float*)d_in[35];
  const float* f2w = (const float*)d_in[36]; const float* f2b = (const float*)d_in[37];
  const float* f2g = (const float*)d_in[38]; const float* f2bb= (const float*)d_in[39];
  const float* f3w = (const float*)d_in[40]; const float* f3b = (const float*)d_in[41];
  const int* src = ei;
  const int* dst = ei + N_EDGES;

  char* wsp = (char*)d_ws;
  size_t off = 0;
  auto carve = [&](size_t bytes) -> char* {
    char* p = wsp + off;
    off = (off + bytes + 255) & ~(size_t)255;
    return p;
  };
  unsigned short* h_bf = (unsigned short*)carve((size_t)N_EDGES * 64 * 2);
  unsigned short* Wp   = (unsigned short*)carve((size_t)256 * 1024 * 2);
  float* s    = (float*)carve((size_t)N_NODES * 64 * 4);
  float* cnt  = (float*)carve((size_t)N_NODES * 4);
  float* pre  = (float*)carve((size_t)N_NODES * 64 * 4);
  float* x1   = (float*)carve((size_t)N_NODES * 64 * 4);
  float* x2   = (float*)carve((size_t)N_NODES * 64 * 4);
  float* x3   = (float*)carve((size_t)N_NODES * 64 * 4);
  float* stats= (float*)carve(2 * 128 * 4);
  float* gate = (float*)carve((size_t)N_NODES * 4);
  float* eexp = (float*)carve((size_t)N_NODES * 4);
  unsigned* mu= (unsigned*)carve(GNUM * 4);
  float* z    = (float*)carve(GNUM * 4);
  float* p    = (float*)carve(GNUM * 64 * 4);
  float* h1   = (float*)carve(GNUM * 128 * 4);
  float* h2   = (float*)carve(GNUM * 64 * 4);
  (void)in_sizes; (void)n_in; (void)out_size; (void)ws_size;

  hipMemsetAsync(cnt, 0, (size_t)N_NODES * 4, stream);
  count_kernel<<<N_EDGES / 256, 256, 0, stream>>>(dst, cnt);

  const float* xin[3] = {x, x1, x2};
  const float* res[3] = {nullptr, x1, x2};
  float* xout[3] = {x1, x2, x3};
  const int inch[3] = {FIN, HDIM, HDIM};

  for (int L = 0; L < 3; ++L) {
    const int NT = inch[L] * 4;
    edge_mlp_kernel<<<(N_EDGES * 64) / 256, 256, 0, stream>>>(ea, ew1[L], eb1[L], h_bf);
    pack_w2_kernel<<<(NT * 1024) / 256, 256, 0, stream>>>(ew2[L], Wp, NT);
    hipMemsetAsync(s, 0, (size_t)N_NODES * 64 * 4, stream);
    nnconv_edge_kernel<<<(N_EDGES / 16) / EW, 32 * EW, 0, stream>>>(
        h_bf, Wp, eb2[L], xin[L], src, dst, s, inch[L], NT);
    finalize_kernel<<<(N_NODES * 64) / 256, 256, 0, stream>>>(
        s, cnt, xin[L], rw[L], rb[L], pre, inch[L]);
    hipMemsetAsync(stats, 0, 2 * 64 * 4, stream);
    bn_stats_kernel<<<64, 256, 0, stream>>>(pre, stats, 64, N_NODES);
    bn_apply_kernel<<<(N_NODES * 64) / 256, 256, 0, stream>>>(
        pre, stats, bg[L], bb[L], res[L], xout[L], 64, N_NODES);
  }

  hipMemsetAsync(mu, 0, GNUM * 4, stream);
  hipMemsetAsync(z, 0, GNUM * 4, stream);
  hipMemsetAsync(p, 0, (size_t)GNUM * 64 * 4, stream);
  gate_kernel<<<N_NODES / 8, 256, 0, stream>>>(x3, gw1, gb1, gw2, gb2, gate, mu, bat);
  attn_exp_kernel<<<N_NODES / 256, 256, 0, stream>>>(gate, mu, bat, eexp, z);
  attn_pool_kernel<<<(N_NODES * 64) / 256, 256, 0, stream>>>(x3, eexp, z, bat, p);

  head_fc_bn_relu<<<1, 128, 0, stream>>>(p, f1w, f1b, f1g, f1bb, h1, 64, 128);
  head_fc_bn_relu<<<1, 64, 0, stream>>>(h1, f2w, f2b, f2g, f2bb, h2, 128, 64);
  head_fc3<<<1, 64, 0, stream>>>(h2, f3w, f3b, (float*)d_out);
}
